// Attention_7550552506968
// MI455X (gfx1250) — matrix operand order repaired and checked by reading, not compiled, not run on hardware
//
#include <hip/hip_runtime.h>
#include <stdint.h>

typedef __attribute__((ext_vector_type(16))) _Float16 v16h;
typedef __attribute__((ext_vector_type(8)))  _Float16 v8h;
typedef __attribute__((ext_vector_type(4)))  _Float16 v4h;
typedef __attribute__((ext_vector_type(8)))  float    v8f;

#define B_   16
#define LQ_  2048
#define LK_  2048
#define D_   512
#define QTILE 32
#define SCALE 0.04419417382415922f   // 1/sqrt(512)
#define NEG_INF (-__builtin_inff())

// ---- WMMA fragment helpers (wave32, v_wmma_f32_16x16x32_f16) --------------
// A-matrix 16x32 f16: lane L holds row M = L&15.
//   lanes 0-15 : halves 0..15 -> K=off0+0..15
//   lanes 16-31: halves 0..15 -> K=off0+16..31 (off0 includes hi*16)
// caller passes off0 = d + hi*16; groups live at off0 and off0+8 (2x16B loads).
__device__ inline v16h frag_a_f16(const _Float16* __restrict__ rowp, int off0) {
  v8h lo = *(const v8h*)(rowp + off0);
  v8h hh = *(const v8h*)(rowp + off0 + 8);
  v16h a;
#pragma unroll
  for (int i = 0; i < 8; ++i) { a[i] = lo[i]; a[8 + i] = hh[i]; }
  return a;
}

// A-fragment built from f32 data (phase 3: P rows come from global weights).
__device__ inline v16h frag_a_f32(const float* __restrict__ rowp, int off0) {
  const float4* p0 = (const float4*)(rowp + off0);
  const float4* p1 = (const float4*)(rowp + off0 + 8);
  float4 x0 = p0[0], x1 = p0[1], y0 = p1[0], y1 = p1[1];
  v16h a;
  a[0]=(_Float16)x0.x; a[1]=(_Float16)x0.y; a[2]=(_Float16)x0.z; a[3]=(_Float16)x0.w;
  a[4]=(_Float16)x1.x; a[5]=(_Float16)x1.y; a[6]=(_Float16)x1.z; a[7]=(_Float16)x1.w;
  a[8]=(_Float16)y0.x; a[9]=(_Float16)y0.y; a[10]=(_Float16)y0.z; a[11]=(_Float16)y0.w;
  a[12]=(_Float16)y1.x; a[13]=(_Float16)y1.y; a[14]=(_Float16)y1.z; a[15]=(_Float16)y1.w;
  return a;
}

// ---- prep kernels ---------------------------------------------------------
// straight f32 -> f16 cast (Q, K), 4 elements/thread, fully coalesced
__global__ __launch_bounds__(256) void cast_f16_kernel(
    const float* __restrict__ src, _Float16* __restrict__ dst) {
  const size_t i = (size_t)blockIdx.x * blockDim.x + threadIdx.x; // one float4
  float4 x = ((const float4*)src)[i];
  v4h h;
  h[0]=(_Float16)x.x; h[1]=(_Float16)x.y; h[2]=(_Float16)x.z; h[3]=(_Float16)x.w;
  ((v4h*)dst)[i] = h;
}

// V (B,LK,D) f32  ->  Vt (B,D,LK) f16 tiled transpose
__global__ __launch_bounds__(256) void transpose_v_kernel(
    const float* __restrict__ v, _Float16* __restrict__ vt) {
  __shared__ float tile[32][33];
  const int b  = blockIdx.z;
  const int k0 = blockIdx.x * 32;
  const int d0 = blockIdx.y * 32;
  const int tx = threadIdx.x;         // 32
  const int ty = threadIdx.y;         // 8
  const float*  vb  = v  + (size_t)b * LK_ * D_;
  _Float16*     vtb = vt + (size_t)b * D_  * LK_;
#pragma unroll
  for (int j = 0; j < 4; ++j) {
    int r = ty + j * 8;
    tile[r][tx] = vb[(size_t)(k0 + r) * D_ + d0 + tx];              // coalesced in d
  }
  __syncthreads();
#pragma unroll
  for (int j = 0; j < 4; ++j) {
    int r = ty + j * 8;
    vtb[(size_t)(d0 + r) * LK_ + k0 + tx] = (_Float16)tile[tx][r];  // coalesced in k
  }
}

// ---- main fused kernel: one block = (batch b, 32 query rows) --------------
__global__ __launch_bounds__(256) void attn_kernel(
    const _Float16* __restrict__ qh,   // (B,LQ,D) f16
    const _Float16* __restrict__ kh,   // (B,LK,D) f16
    const uint8_t*  __restrict__ mask, // (B,LK)
    const _Float16* __restrict__ vt,   // (B,D,LK) f16
    float* __restrict__ out,           // (B,LQ,D)
    float* __restrict__ weights)       // (B,LQ,LK)
{
  const int b    = blockIdx.y;
  const int q0   = blockIdx.x * QTILE;
  const int tid  = threadIdx.x;
  const int w    = tid >> 5;          // wave id 0..7
  const int lane = tid & 31;
  const int m    = lane & 15;
  const int hi   = lane >> 4;         // lane half-select

  const _Float16* Qb  = qh + (size_t)(b * LQ_ + q0) * D_;
  const _Float16* Kb  = kh + (size_t)b * LK_ * D_;
  const uint8_t*  Mb  = mask + (size_t)b * LK_;
  float*          Wb  = weights + (size_t)(b * LQ_ + q0) * LK_;  // 32 x 2048 panel
  float*          Ob  = out + (size_t)(b * LQ_ + q0) * D_;
  const _Float16* Vtb = vt + (size_t)b * D_ * LK_;

  // ---------------- Phase 1: masked scaled logits -> global weights -------
  // wave w owns K-tiles kt in [16w, 16w+16); 4 tiles share each A-fragment.
#pragma unroll
  for (int qi = 0; qi < 2; ++qi) {
    const _Float16* arow = Qb + (size_t)(qi * 16 + m) * D_;
    for (int g = 0; g < 4; ++g) {
      const int ktb = w * 16 + g * 4;
      const _Float16* brow[4];
#pragma unroll
      for (int j = 0; j < 4; ++j)
        brow[j] = Kb + (size_t)((ktb + j) * 16 + m) * D_ + hi * 16;
      v8f acc[4];
#pragma unroll
      for (int j = 0; j < 4; ++j) acc[j] = (v8f){};
#pragma unroll 4
      for (int d = 0; d < D_; d += 32) {
        v16h af = frag_a_f16(arow, d + hi * 16);
#pragma unroll
        for (int j = 0; j < 4; ++j) {
          v16h bf = *(const v16h*)(brow[j] + d);     // 32B contiguous, 32B-aligned
          acc[j] = __builtin_amdgcn_wmma_f32_16x16x32_f16(
              false, af, false, bf, (short)0, acc[j], false, false);
        }
      }
#pragma unroll
      for (int j = 0; j < 4; ++j) {
        const int col = (ktb + j) * 16 + m;
        const bool msk = Mb[col] != 0;
#pragma unroll
        for (int r = 0; r < 8; ++r) {
          const int row = qi * 16 + 8 * hi + r;      // C layout: lanes>=16 hold M+8
          float vv = msk ? NEG_INF : acc[j][r] * SCALE;
          Wb[(size_t)row * LK_ + col] = vv;          // coalesced across lanes 0-15
        }
      }
    }
  }
  __threadfence_block();
  __syncthreads();

  // ---------------- Phase 2: row softmax in place (L2-resident panel) -----
#pragma unroll
  for (int i = 0; i < 4; ++i) {
    float* wr = Wb + (size_t)(w * 4 + i) * LK_;
    float mx = NEG_INF;
    for (int c = lane; c < LK_; c += 32) mx = fmaxf(mx, wr[c]);
#pragma unroll
    for (int off = 16; off > 0; off >>= 1) mx = fmaxf(mx, __shfl_xor(mx, off, 32));
    float s = 0.0f;
    for (int c = lane; c < LK_; c += 32) {
      float e = __expf(wr[c] - mx);                  // -inf -> 0 for masked cols
      wr[c] = e;
      s += e;
    }
#pragma unroll
    for (int off = 16; off > 0; off >>= 1) s += __shfl_xor(s, off, 32);
    const float inv = 1.0f / s;                      // row never fully masked
    for (int c = lane; c < LK_; c += 32) wr[c] *= inv;
  }
  __threadfence_block();
  __syncthreads();

  // ---------------- Phase 3: O = P @ V (f16 WMMA, f32 accum) --------------
  // wave w owns output columns [vt0*16, (vt0+4)*16), both q sub-tiles.
  const int vt0 = w * 4;
  v8f acc[2][4];
#pragma unroll
  for (int qi = 0; qi < 2; ++qi)
#pragma unroll
    for (int j = 0; j < 4; ++j) acc[qi][j] = (v8f){};

  const float* prow0 = Wb + (size_t)m * LK_;
  const float* prow1 = Wb + (size_t)(16 + m) * LK_;

  for (int kc = 0; kc < LK_; kc += 32) {
    v16h bh[4];
#pragma unroll
    for (int j = 0; j < 4; ++j) {
      const int vcol = (vt0 + j) * 16 + m;
      bh[j] = *(const v16h*)(Vtb + (size_t)vcol * LK_ + kc + hi * 16); // 32B load
    }
#pragma unroll
    for (int qi = 0; qi < 2; ++qi) {
      v16h af = frag_a_f32(qi ? prow1 : prow0, kc + hi * 16); // 2 cvt_pk per WMMA
#pragma unroll
      for (int j = 0; j < 4; ++j)
        acc[qi][j] = __builtin_amdgcn_wmma_f32_16x16x32_f16(
            false, af, false, bh[j], (short)0, acc[qi][j], false, false);
    }
  }
#pragma unroll
  for (int qi = 0; qi < 2; ++qi)
#pragma unroll
    for (int j = 0; j < 4; ++j)
#pragma unroll
      for (int r = 0; r < 8; ++r) {
        const int row = qi * 16 + 8 * hi + r;
        Ob[(size_t)row * D_ + (vt0 + j) * 16 + m] = acc[qi][j][r];
      }
}

// ---------------------------------------------------------------------------
extern "C" void kernel_launch(void* const* d_in, const int* in_sizes, int n_in,
                              void* d_out, int out_size, void* d_ws, size_t ws_size,
                              hipStream_t stream) {
  (void)in_sizes; (void)n_in; (void)out_size; (void)ws_size;
  const float*   q    = (const float*)d_in[0];
  const float*   k    = (const float*)d_in[1];
  const float*   v    = (const float*)d_in[2];
  const uint8_t* mask = (const uint8_t*)d_in[3];

  float* out     = (float*)d_out;                          // (B,LQ,D) first
  float* weights = out + (size_t)B_ * LQ_ * D_;            // then (B,LQ,LK)

  const size_t nElem = (size_t)B_ * LQ_ * D_;              // == B*LK*D == 16.7M
  _Float16* qh = (_Float16*)d_ws;                          // 32 MiB
  _Float16* kh = qh + nElem;                               // 32 MiB
  _Float16* vt = kh + nElem;                               // 32 MiB

  const int castBlocks = (int)(nElem / 4 / 256);           // 4 elems per thread
  cast_f16_kernel<<<castBlocks, 256, 0, stream>>>(q, qh);
  cast_f16_kernel<<<castBlocks, 256, 0, stream>>>(k, kh);

  dim3 tgrid(LK_ / 32, D_ / 32, B_);
  transpose_v_kernel<<<tgrid, dim3(32, 8), 0, stream>>>(v, vt);

  dim3 agrid(LQ_ / QTILE, B_);
  attn_kernel<<<agrid, 256, 0, stream>>>(qh, kh, mask, vt, out, weights);
}
